// MultiHeadAttention_9655086482251
// MI455X (gfx1250) — compile-verified, matrix operand order checked by reading, not run on hardware
//
#include <hip/hip_runtime.h>

// ---------------------------------------------------------------------------
// MI455X (gfx1250) multi-head attention.
// f16 WMMA (f32 accum), async-to-LDS double-buffered GEMM, causal flash attn.
// ---------------------------------------------------------------------------

typedef __attribute__((ext_vector_type(16))) _Float16 v16h;
typedef __attribute__((ext_vector_type(8)))  _Float16 v8h;
typedef __attribute__((ext_vector_type(8)))  float    v8f;

#define LOG2E 1.44269504088896340f

__device__ __forceinline__ v8f wmma16(v16h a, v16h b, v8f c) {
  // D = A(16x32 f16) * B(32x16 f16) + C(16x16 f32)
  return __builtin_amdgcn_wmma_f32_16x16x32_f16(
      /*neg_a=*/false, a, /*neg_b=*/false, b,
      /*c_mod=*/(short)0, c, /*reuse_a=*/false, /*reuse_b=*/false);
}

// Load a 16-bit WMMA fragment from row-major storage.
// Caller passes p = base + (lane&15)*stride + ((lane>>4)<<3).
// CDNA5 layout: halves 0..7 = k {ko..ko+7}, halves 8..15 = k {ko+16..ko+23}.
__device__ __forceinline__ v16h frag_ld(const _Float16* p) {
  v8h lo = *(const v8h*)p;
  v8h hi = *(const v8h*)(p + 16);
  v16h r;
#pragma unroll
  for (int i = 0; i < 8; ++i) { r[i] = lo[i]; r[i + 8] = hi[i]; }
  return r;
}

// CDNA5 async copy: 16 bytes global -> LDS per lane, tracked by ASYNCcnt.
// lds_addr is the workgroup-relative LDS byte offset (low 32 bits of the
// generic pointer to a __shared__ object).
__device__ __forceinline__ void async_ld_b128(unsigned lds_addr, const void* g) {
  unsigned long long ga = (unsigned long long)(size_t)g;
  asm volatile("global_load_async_to_lds_b128 %0, %1, off"
               :: "v"(lds_addr), "v"(ga) : "memory");
}
__device__ __forceinline__ void wait_async0() {
  asm volatile("s_wait_asynccnt 0x0" ::: "memory");
}

// ---------------------------------------------------------------------------
// f32 -> f16 conversion pass (one-time; whole problem then lives in L2 as f16)
// ---------------------------------------------------------------------------
__global__ __launch_bounds__(256) void cvt_f16(const float* __restrict__ in,
                                               _Float16* __restrict__ out) {
  size_t i = ((size_t)blockIdx.x * 256 + threadIdx.x) * 8;
  float4 a = *(const float4*)(in + i);
  float4 b = *(const float4*)(in + i + 4);
  v8h o;
  o[0] = (_Float16)a.x; o[1] = (_Float16)a.y;
  o[2] = (_Float16)a.z; o[3] = (_Float16)a.w;
  o[4] = (_Float16)b.x; o[5] = (_Float16)b.y;
  o[6] = (_Float16)b.z; o[7] = (_Float16)b.w;
  *(v8h*)(out + i) = o;
}

// ---------------------------------------------------------------------------
// Projection GEMM (all-f16 inputs): Out[M,N] = A[M,K] @ W[N,K]^T + bias
//   SPLIT_HEADS: write f16 [B=2,H=16,S=2048,Dh=64]; else f32 [M,N]
// WG tile 128x128, BK=32, 8 waves (4x2), wave tile 32x64 (8 WMMA / K-step).
// LDS tiles double-buffered, filled with global_load_async_to_lds_b128.
// ---------------------------------------------------------------------------
template <bool SPLIT_HEADS>
__global__ __launch_bounds__(256) void gemm_f16(
    const _Float16* __restrict__ A, const _Float16* __restrict__ W,
    const float* __restrict__ bias, void* __restrict__ Out,
    int M, int N, int K) {
  __shared__ alignas(16) _Float16 As[2][128][40];
  __shared__ alignas(16) _Float16 Bs[2][128][40];

  const int tid  = threadIdx.x;
  const int lane = tid & 31;
  const int wid  = tid >> 5;        // 0..7
  const int wm   = wid & 3;         // 4 waves along M
  const int wn   = wid >> 2;        // 2 waves along N
  const int nbase = blockIdx.x * 128;
  const int mbase = blockIdx.y * 128;

  const int fm = lane & 15;
  const int fk = (lane >> 4) << 3;

  const unsigned asBase = (unsigned)(size_t)&As[0][0][0];
  const unsigned bsBase = (unsigned)(size_t)&Bs[0][0][0];

  // Stage one 128x32 f16 tile of A and of W into LDS buffer `buf` (async).
  // 512 16B chunks per tile -> 2 per thread per tile.
  auto stage = [&](int buf, int k0) {
#pragma unroll
    for (int i = 0; i < 2; ++i) {
      int cid = tid + i * 256;
      int row = cid >> 2, c16 = (cid & 3) << 3;      // 8-half chunks
      unsigned loff = (unsigned)(buf * 10240 + row * 80 + c16 * 2);
      async_ld_b128(asBase + loff, A + (size_t)(mbase + row) * K + k0 + c16);
      async_ld_b128(bsBase + loff, W + (size_t)(nbase + row) * K + k0 + c16);
    }
  };

  v8f acc[2][4] = {};

  stage(0, 0);
  int buf = 0;
  for (int k0 = 0; k0 < K; k0 += 32) {
    wait_async0();       // my fills for `buf` have landed
    __syncthreads();     // everyone's fills have landed / prev reads done
    if (k0 + 32 < K) stage(buf ^ 1, k0 + 32);   // overlap next fill w/ WMMAs

    v16h a0 = frag_ld(&As[buf][wm * 32      + fm][fk]);
    v16h a1 = frag_ld(&As[buf][wm * 32 + 16 + fm][fk]);
#pragma unroll
    for (int j = 0; j < 4; ++j) {
      v16h bj = frag_ld(&Bs[buf][wn * 64 + j * 16 + fm][fk]);
      acc[0][j] = wmma16(a0, bj, acc[0][j]);
      acc[1][j] = wmma16(a1, bj, acc[1][j]);
    }
    buf ^= 1;
  }

  // Epilogue: C layout -> lane col = lane&15; VGPR r holds row (lane>=16?8:0)+r
#pragma unroll
  for (int i = 0; i < 2; ++i) {
#pragma unroll
    for (int j = 0; j < 4; ++j) {
      int gn = nbase + wn * 64 + j * 16 + fm;
      float bval = bias[gn];
#pragma unroll
      for (int r = 0; r < 8; ++r) {
        int gm = mbase + wm * 32 + i * 16 + fk + r;
        float val = acc[i][j][r] + bval;
        if (SPLIT_HEADS) {
          int bI = gm >> 11, s = gm & 2047, hI = gn >> 6, d = gn & 63;
          ((_Float16*)Out)[(((size_t)(bI * 16 + hI) * 2048 + s)) * 64 + d] =
              (_Float16)val;
        } else {
          ((float*)Out)[(size_t)gm * N + gn] = val;
        }
      }
    }
  }
}

// ---------------------------------------------------------------------------
// Causal flash attention. One WG per (b, h, 128-query block); 8 waves, each
// owning 16 query rows. 32-key blocks. K block async-staged to LDS once per
// WG; V staged transposed; P converted C-layout -> A-fragment via per-wave
// LDS scratch. Masking runs only on wave-uniform diagonal (partial) blocks.
// ---------------------------------------------------------------------------
__global__ __launch_bounds__(256) void flash_attn(
    const _Float16* __restrict__ Qh, const _Float16* __restrict__ Kh,
    const _Float16* __restrict__ Vh, _Float16* __restrict__ Ao) {
  __shared__ alignas(16) _Float16 Ks[32][72];      // K block, row-major
  __shared__ alignas(16) _Float16 Vs[64][40];      // V^T block: [d][t]
  __shared__ alignas(16) _Float16 Ps[8][16][40];   // per-wave P scratch

  const int tid  = threadIdx.x;
  const int lane = tid & 31;
  const int wid  = tid >> 5;
  const int b = blockIdx.z, h = blockIdx.y;
  const int qbase = blockIdx.x * 128;
  const int qr0 = qbase + wid * 16;

  const size_t bh = (size_t)(b * 16 + h) * 2048 * 64;
  const _Float16* Qb = Qh + bh;
  const _Float16* Kb = Kh + bh;
  const _Float16* Vb = Vh + bh;

  const int fm = lane & 15;
  const int fk = (lane >> 4) << 3;
  const unsigned ksBase = (unsigned)(size_t)&Ks[0][0];

  // Persistent Q fragments: rows qr0..qr0+15, d = 0..31 / 32..63.
  const _Float16* qp = Qb + (size_t)(qr0 + fm) * 64 + fk;
  v16h aq0 = frag_ld(qp);
  v16h aq1 = frag_ld(qp + 32);

  v8f accv[4] = {};
  float mrow[8], lrow[8];
#pragma unroll
  for (int r = 0; r < 8; ++r) { mrow[r] = -1e30f; lrow[r] = 0.f; }

  const float SC = 0.125f * LOG2E;   // 1/sqrt(Dh) folded with log2(e)
  const int kend = qbase + 128;      // causal extent of this query block

  for (int kb = 0; kb < kend; kb += 32) {
    __syncthreads();                 // prev reads of Ks/Vs complete
    {  // async-stage K[kb..kb+31][0..63]: one b128 chunk per thread
      int row = tid >> 3, c0 = (tid & 7) << 3;
      async_ld_b128(ksBase + (unsigned)(row * 144 + c0 * 2),
                    Kb + (size_t)(kb + row) * 64 + c0);
    }
    {  // stage V block transposed (register transpose -> ds stores)
      int vr = tid >> 3, c0 = (tid & 7) << 3;
      v8h vv = *(const v8h*)(Vb + (size_t)(kb + vr) * 64 + c0);
#pragma unroll
      for (int i = 0; i < 8; ++i) Vs[c0 + i][vr] = vv[i];
    }
    wait_async0();
    __syncthreads();
    if (kb > qr0 + 15) continue;     // wave-uniform: no visible keys here

    // S = Q K^T : two 16x16 key tiles, k-dim = Dh in two steps of 32
    v8f s0 = {}, s1 = {};
    s0 = wmma16(aq0, frag_ld(&Ks[fm][fk]), s0);
    s0 = wmma16(aq1, frag_ld(&Ks[fm][fk + 32]), s0);
    s1 = wmma16(aq0, frag_ld(&Ks[16 + fm][fk]), s1);
    s1 = wmma16(aq1, frag_ld(&Ks[16 + fm][fk + 32]), s1);

#pragma unroll
    for (int r = 0; r < 8; ++r) { s0[r] *= SC; s1[r] *= SC; }

    if (kb + 31 > qr0) {             // diagonal block(s) only: causal mask
      const int col0 = kb + fm, col1 = col0 + 16;
      const int rb = qr0 + fk;
#pragma unroll
      for (int r = 0; r < 8; ++r) {
        if (col0 > rb + r) s0[r] = -1e30f;
        if (col1 > rb + r) s1[r] = -1e30f;
      }
    }

#pragma unroll
    for (int r = 0; r < 8; ++r) {
      float mx = fmaxf(s0[r], s1[r]);
#pragma unroll
      for (int off = 8; off >= 1; off >>= 1)
        mx = fmaxf(mx, __shfl_xor(mx, off, 32));
      float mnew = fmaxf(mrow[r], mx);
      float corr = exp2f(mrow[r] - mnew);
      float p0 = exp2f(s0[r] - mnew);
      float p1 = exp2f(s1[r] - mnew);
      float ps = p0 + p1;
#pragma unroll
      for (int off = 8; off >= 1; off >>= 1)
        ps += __shfl_xor(ps, off, 32);
      lrow[r] = lrow[r] * corr + ps;
      mrow[r] = mnew;
      accv[0][r] *= corr; accv[1][r] *= corr;
      accv[2][r] *= corr; accv[3][r] *= corr;
      s0[r] = p0; s1[r] = p1;
    }

    // P: C-layout -> A-fragment via per-wave LDS scratch (wave-local, DS in order)
    _Float16* Pw = &Ps[wid][0][0];
#pragma unroll
    for (int r = 0; r < 8; ++r) {
      Pw[(fk + r) * 40 + fm]      = (_Float16)s0[r];
      Pw[(fk + r) * 40 + 16 + fm] = (_Float16)s1[r];
    }
    v16h ap = frag_ld(&Ps[wid][fm][fk]);

    // O += P @ V : four 16-wide dh tiles from transposed Vs
#pragma unroll
    for (int nt = 0; nt < 4; ++nt) {
      v16h bvv = frag_ld(&Vs[nt * 16 + fm][fk]);
      accv[nt] = wmma16(ap, bvv, accv[nt]);
    }
  }

  // Epilogue: normalize, store merged-head f16 [B, S, 1024]
#pragma unroll
  for (int nt = 0; nt < 4; ++nt) {
    int gcol = h * 64 + nt * 16 + fm;
#pragma unroll
    for (int r = 0; r < 8; ++r) {
      int grow = qr0 + fk + r;
      Ao[((size_t)(b * 2048 + grow)) * 1024 + gcol] =
          (_Float16)(accv[nt][r] / lrow[r]);
    }
  }
}

// ---------------------------------------------------------------------------
extern "C" void kernel_launch(void* const* d_in, const int* in_sizes, int n_in,
                              void* d_out, int out_size, void* d_ws,
                              size_t ws_size, hipStream_t stream) {
  (void)in_sizes; (void)n_in; (void)out_size; (void)ws_size;
  const float* v  = (const float*)d_in[0];
  const float* k  = (const float*)d_in[1];
  const float* q  = (const float*)d_in[2];
  // d_in[3] (dense causal mask) intentionally unused: handled analytically.
  const float* Wq = (const float*)d_in[4];
  const float* bq = (const float*)d_in[5];
  const float* Wk = (const float*)d_in[6];
  const float* bk = (const float*)d_in[7];
  const float* Wv = (const float*)d_in[8];
  const float* bv = (const float*)d_in[9];
  const float* Wo = (const float*)d_in[10];
  const float* bo = (const float*)d_in[11];

  const size_t NACT = (size_t)4096 * 1024;   // activation elements
  const size_t NW   = (size_t)1024 * 1024;   // weight elements
  _Float16* p = (_Float16*)d_ws;
  _Float16 *qf = p, *kf = p + NACT, *vf = p + 2 * NACT;   p += 3 * NACT;
  _Float16 *Wqf = p, *Wkf = p + NW, *Wvf = p + 2 * NW, *Wof = p + 3 * NW;
  p += 4 * NW;
  _Float16 *QhB = p, *KhB = p + NACT, *VhB = p + 2 * NACT; p += 3 * NACT;
  _Float16 *Aoh = p;                                        // [B,S,D] f16

  dim3 blk(256);
  // One-time f32 -> f16 conversion (working set then f16-resident in L2).
  cvt_f16<<<NACT / 2048, blk, 0, stream>>>(q, qf);
  cvt_f16<<<NACT / 2048, blk, 0, stream>>>(k, kf);
  cvt_f16<<<NACT / 2048, blk, 0, stream>>>(v, vf);
  cvt_f16<<<NW / 2048, blk, 0, stream>>>(Wq, Wqf);
  cvt_f16<<<NW / 2048, blk, 0, stream>>>(Wk, Wkf);
  cvt_f16<<<NW / 2048, blk, 0, stream>>>(Wv, Wvf);
  cvt_f16<<<NW / 2048, blk, 0, stream>>>(Wo, Wof);

  dim3 gp(1024 / 128, 4096 / 128);           // (N tiles, M tiles)
  gemm_f16<true><<<gp, blk, 0, stream>>>(qf, Wqf, bq, QhB, 4096, 1024, 1024);
  gemm_f16<true><<<gp, blk, 0, stream>>>(kf, Wkf, bk, KhB, 4096, 1024, 1024);
  gemm_f16<true><<<gp, blk, 0, stream>>>(vf, Wvf, bv, VhB, 4096, 1024, 1024);
  flash_attn<<<dim3(16, 16, 2), blk, 0, stream>>>(QhB, KhB, VhB, Aoh);
  gemm_f16<false><<<gp, blk, 0, stream>>>(Aoh, Wof, bo, d_out, 4096, 1024, 1024);
}
